// CausalSelfAttention_12283606468077
// MI455X (gfx1250) — matrix operand order repaired and checked by reading, not compiled, not run on hardware
//
#include <hip/hip_runtime.h>

typedef __bf16 bf16;
typedef __attribute__((ext_vector_type(16))) __bf16 v16bf;
typedef __attribute__((ext_vector_type(8)))  float  v8f;

struct __align__(16) U4 { unsigned int x, y, z, w; };

// ---------------------------------------------------------- TDM availability
#if defined(__has_builtin)
#if __has_builtin(__builtin_amdgcn_tensor_load_to_lds) && __has_builtin(__builtin_amdgcn_s_wait_tensorcnt)
#define USE_TDM 1
#endif
#endif
#ifndef USE_TDM
#define USE_TDM 0
#endif

#if USE_TDM
typedef __attribute__((ext_vector_type(4))) unsigned int u32x4;
typedef __attribute__((ext_vector_type(8))) int          i32x8;
typedef __attribute__((ext_vector_type(4))) int          i32x4;

// 2D bf16 tile load: tile (d1 rows x d0 elems), row stride in elements.
// D# packing per CDNA5 ISA (count=1, type=2,
// data_size=2B, workgroup_mask=0, tensor dims == tile dims -> fully in-bounds).
__device__ __forceinline__ void tdm_load_2d(unsigned lds_addr, const void* gaddr,
                                            unsigned d0, unsigned d1,
                                            unsigned long long stride_elems) {
    unsigned long long ga = (unsigned long long)(size_t)gaddr;
    u32x4 g0;
    g0[0] = 1u;                                             // count=1
    g0[1] = lds_addr;                                       // LDS byte address
    g0[2] = (unsigned)(ga & 0xFFFFFFFFu);                   // global_addr[31:0]
    g0[3] = (unsigned)((ga >> 32) & 0x1FFFFFFu) | (2u << 30); // [56:32] | type=2
    i32x8 g1;
    g1[0] = (int)(1u << 16);                                // data_size=1 (2B)
    g1[1] = (int)((d0 & 0xFFFFu) << 16);                    // tensor_dim0[15:0]
    g1[2] = (int)(((d0 >> 16) & 0xFFFFu) | ((d1 & 0xFFFFu) << 16)); // td0 hi | td1 lo
    g1[3] = (int)(((d1 >> 16) & 0xFFFFu) | ((d0 & 0xFFFFu) << 16)); // td1 hi | tile_dim0
    g1[4] = (int)(d1 & 0xFFFFu);                            // tile_dim1 (tile_dim2=0)
    g1[5] = (int)(unsigned)(stride_elems & 0xFFFFFFFFull);  // dim0_stride[31:0]
    g1[6] = (int)(unsigned)((stride_elems >> 32) & 0xFFFFull); // dim0_stride[47:32]
    g1[7] = 0;
    i32x4 z4 = {0, 0, 0, 0};
    i32x8 z8 = {0, 0, 0, 0, 0, 0, 0, 0};
    __builtin_amdgcn_tensor_load_to_lds(g0, g1, z4, z4, z8, 0);
}
__device__ __forceinline__ unsigned lds_off(const void* p) {
    return (unsigned)(size_t)p;   // generic LDS addr: low 32 bits = LDS offset
}
#endif

// ---------------------------------------------------------------- f32 -> bf16
__global__ __launch_bounds__(256)
void cvt_f32_bf16(const float* __restrict__ src, bf16* __restrict__ dst, long n) {
    long i = (long)blockIdx.x * blockDim.x + threadIdx.x;
    long stride = (long)gridDim.x * blockDim.x;
    for (; i < n; i += stride) dst[i] = (bf16)src[i];
}

// --------------------------------------------------------- bf16 WMMA GEMM
// C(f32, MxN) = A(bf16, MxK row-major) * B(bf16, KxN row-major)
// block tile 128x128, BK=32, 256 threads = 8 waves (4 M x 2 N), wave tile 32x64.
// Double-buffered LDS; A tile via TDM, B tile register-staged + transposed.
template <int N, int K>
__global__ __launch_bounds__(256)
void gemm_bf16_wmma(const bf16* __restrict__ A, const bf16* __restrict__ B,
                    float* __restrict__ C, int M) {
    __shared__ __align__(16) bf16 As[2][128 * 32];   // [m][k]
    __shared__ __align__(16) bf16 Bs[2][128 * 32];   // transposed: [n][k]

    const int tid    = threadIdx.x;
    const int wave   = tid >> 5;
    const int lane   = tid & 31;
    const int lanelo = lane & 15;
    const int hi     = lane >> 4;
    const int m0 = blockIdx.y * 128;
    const int n0 = blockIdx.x * 128;
    const int wm = (wave >> 1) * 32;
    const int wn = (wave & 1) * 64;

#if !USE_TDM
    const int arow0 = tid >> 2,         akc0 = (tid & 3) * 8;
    const int arow1 = (tid + 256) >> 2, akc1 = (tid & 3) * 8;
    U4 ar0, ar1;
    auto loadA = [&](int kk) {
        ar0 = *reinterpret_cast<const U4*>(&A[(long)(m0 + arow0) * K + kk + akc0]);
        ar1 = *reinterpret_cast<const U4*>(&A[(long)(m0 + arow1) * K + kk + akc1]);
    };
    auto storeA = [&](int buf) {
        *reinterpret_cast<U4*>(&As[buf][arow0 * 32 + akc0]) = ar0;
        *reinterpret_cast<U4*>(&As[buf][arow1 * 32 + akc1]) = ar1;
    };
#endif
    // B staging: one (n-chunk, k-pair) per thread: rows 2kp, 2kp+1
    const int bkp = tid >> 4, bnc = (tid & 15) * 8;
    U4 br0, br1;
    auto loadB = [&](int kk) {
        br0 = *reinterpret_cast<const U4*>(&B[(long)(kk + 2 * bkp)     * N + n0 + bnc]);
        br1 = *reinterpret_cast<const U4*>(&B[(long)(kk + 2 * bkp + 1) * N + n0 + bnc]);
    };
    auto storeB = [&](int buf) {
        const unsigned short* t0 = reinterpret_cast<const unsigned short*>(&br0);
        const unsigned short* t1 = reinterpret_cast<const unsigned short*>(&br1);
        #pragma unroll
        for (int j = 0; j < 8; ++j)
            *reinterpret_cast<unsigned int*>(&Bs[buf][(bnc + j) * 32 + 2 * bkp]) =
                (unsigned)t0[j] | ((unsigned)t1[j] << 16);
    };

    v8f acc[2][4] = {};

    // prologue: stage tile 0
#if USE_TDM
    if (wave == 0)
        tdm_load_2d(lds_off(&As[0][0]), &A[(long)m0 * K], 32, 128, (unsigned long long)K);
#else
    loadA(0);
    storeA(0);
#endif
    loadB(0);
    storeB(0);
    int cur = 0;

    #pragma unroll 1
    for (int kk = 0; kk < K; kk += 32) {
#if USE_TDM
        __builtin_amdgcn_s_wait_tensorcnt((short)0);  // drain before handing tile over
#endif
        __syncthreads();
        const bool has_next = (kk + 32 < K);
        if (has_next) {
#if USE_TDM
            if (wave == 0)
                tdm_load_2d(lds_off(&As[cur ^ 1][0]), &A[(long)m0 * K + kk + 32],
                            32, 128, (unsigned long long)K);
#else
            loadA(kk + 32);
#endif
            loadB(kk + 32);                           // overlaps WMMA below
        }

        v16bf a[2], b[4];
        #pragma unroll
        for (int mi = 0; mi < 2; ++mi) {
            const bf16* p = &As[cur][(wm + mi * 16 + lanelo) * 32 + hi * 8];
            reinterpret_cast<U4*>(&a[mi])[0] = *reinterpret_cast<const U4*>(p);
            reinterpret_cast<U4*>(&a[mi])[1] = *reinterpret_cast<const U4*>(p + 16);
        }
        #pragma unroll
        for (int ni = 0; ni < 4; ++ni) {
            const bf16* p = &Bs[cur][(wn + ni * 16 + lanelo) * 32 + hi * 8];
            reinterpret_cast<U4*>(&b[ni])[0] = *reinterpret_cast<const U4*>(p);
            reinterpret_cast<U4*>(&b[ni])[1] = *reinterpret_cast<const U4*>(p + 16);
        }
        #pragma unroll
        for (int mi = 0; mi < 2; ++mi)
            #pragma unroll
            for (int ni = 0; ni < 4; ++ni)
                acc[mi][ni] = __builtin_amdgcn_wmma_f32_16x16x32_bf16(
                    false, a[mi], false, b[ni], (short)0, acc[mi][ni], false, false);

        if (has_next) {
#if !USE_TDM
            storeA(cur ^ 1);
#endif
            storeB(cur ^ 1);
            cur ^= 1;
        }
    }

    #pragma unroll
    for (int mi = 0; mi < 2; ++mi)
        #pragma unroll
        for (int ni = 0; ni < 4; ++ni)
            #pragma unroll
            for (int r = 0; r < 8; ++r) {
                int row = m0 + wm + mi * 16 + hi * 8 + r;
                int col = n0 + wn + ni * 16 + lanelo;
                C[(long)row * N + col] = acc[mi][ni][r];
            }
}

// ------------------------------------------------- RoPE + RMSNorm + relayout
// raw: (B,T,H,128) f32, out: (B,H,T,128) bf16. One wave per head-vector,
// pure shfl reduction: no LDS, no barriers.
__global__ __launch_bounds__(32)
void rope_rms(const float* __restrict__ raw, const float* __restrict__ cosp,
              const float* __restrict__ sinp, bf16* __restrict__ out,
              int H, int T) {
    const int i = threadIdx.x;            // 0..31
    const long blk = blockIdx.x;          // ((b*T + t)*H + h)
    const int h = (int)(blk % H);
    const long bt = blk / H;
    const int t = (int)(bt % T);
    const int b = (int)(bt / T);

    const float* src = raw + blk * 128;
    float x1a = src[i],      x1b = src[i + 32];
    float x2a = src[i + 64], x2b = src[i + 96];
    float ca = cosp[t * 64 + i], cb = cosp[t * 64 + i + 32];
    float sa = sinp[t * 64 + i], sb = sinp[t * 64 + i + 32];
    float y1a =  x1a * ca + x2a * sa, y2a = -x1a * sa + x2a * ca;
    float y1b =  x1b * cb + x2b * sb, y2b = -x1b * sb + x2b * cb;

    float ss = y1a * y1a + y2a * y2a + y1b * y1b + y2b * y2b;
    #pragma unroll
    for (int mk = 16; mk >= 1; mk >>= 1) ss += __shfl_xor(ss, mk, 32);
    float r = rsqrtf(ss * (1.0f / 128.0f) + 1e-6f);

    long ob = (((long)b * H + h) * T + t) * 128;
    out[ob + i]      = (bf16)(y1a * r);
    out[ob + i + 32] = (bf16)(y1b * r);
    out[ob + i + 64] = (bf16)(y2a * r);
    out[ob + i + 96] = (bf16)(y2b * r);
}

// V: (B,T,KH,128) f32 -> (B,KH,T,128) bf16
__global__ __launch_bounds__(128)
void v_relayout(const float* __restrict__ raw, bf16* __restrict__ out, int H, int T) {
    const long blk = blockIdx.x;          // ((b*T + t)*H + h)
    const int i = threadIdx.x;
    const int h = (int)(blk % H);
    const long bt = blk / H;
    const int t = (int)(bt % T);
    const int b = (int)(bt / T);
    out[(((long)b * H + h) * T + t) * 128 + i] = (bf16)raw[blk * 128 + i];
}

// ------------------------------------------------------------- flash attention
// grid: (B*16 heads, T/128). block 256 = 8 waves; wave w owns query rows 16w..16w+15.
// Double-buffered 32-key K/V tiles: K via TDM, V register-staged + transposed.
__global__ __launch_bounds__(256)
void attn_flash(const bf16* __restrict__ Q, const bf16* __restrict__ K,
                const bf16* __restrict__ V, bf16* __restrict__ Y) {
    __shared__ __align__(16) bf16 Ks[2][32 * 128];    // [key][d]   (B-frag layout)
    __shared__ __align__(16) bf16 Vt[2][128 * 32];    // [d][key]   (transposed)
    __shared__ __align__(16) bf16 Ps[8 * 16 * 32];    // wave-private P staging

    const int tid    = threadIdx.x;
    const int wave   = tid >> 5;
    const int lane   = tid & 31;
    const int lanelo = lane & 15;
    const int hi     = lane >> 4;
    const int bh = blockIdx.x;            // b*16 + h
    const int b  = bh >> 4, h = bh & 15;
    const int kh = h >> 2;                // nrep = 16/4
    const int q0 = blockIdx.y * 128;
    const long qoff  = (long)bh * 2048 * 128;
    const long kvoff = ((long)b * 4 + kh) * 2048 * 128;
    const float scale = 0.08838834764831845f;  // 1/sqrt(128)

#if !USE_TDM
    const int krow0 = tid >> 4,         kdc0 = (tid & 15) * 8;
    const int krow1 = (tid + 256) >> 4, kdc1 = (tid & 15) * 8;
    U4 kr0, kr1;
    auto loadK = [&](int kst) {
        kr0 = *reinterpret_cast<const U4*>(&K[kvoff + (long)(kst + krow0) * 128 + kdc0]);
        kr1 = *reinterpret_cast<const U4*>(&K[kvoff + (long)(kst + krow1) * 128 + kdc1]);
    };
    auto storeK = [&](int buf) {
        *reinterpret_cast<U4*>(&Ks[buf][krow0 * 128 + kdc0]) = kr0;
        *reinterpret_cast<U4*>(&Ks[buf][krow1 * 128 + kdc1]) = kr1;
    };
#endif
    // V staging: one (d-chunk, t-pair) per thread
    const int vtp = tid >> 4, vdc = (tid & 15) * 8;
    U4 vr0, vr1;
    auto loadV = [&](int kst) {
        vr0 = *reinterpret_cast<const U4*>(&V[kvoff + (long)(kst + 2 * vtp)     * 128 + vdc]);
        vr1 = *reinterpret_cast<const U4*>(&V[kvoff + (long)(kst + 2 * vtp + 1) * 128 + vdc]);
    };
    auto storeV = [&](int buf) {
        const unsigned short* t0 = reinterpret_cast<const unsigned short*>(&vr0);
        const unsigned short* t1 = reinterpret_cast<const unsigned short*>(&vr1);
        #pragma unroll
        for (int j = 0; j < 8; ++j)
            *reinterpret_cast<unsigned int*>(&Vt[buf][(vdc + j) * 32 + 2 * vtp]) =
                (unsigned)t0[j] | ((unsigned)t1[j] << 16);
    };

    // Q fragments straight from global (A-frag layout is contiguous in d)
    v16bf qf[4];
    {
        const int row = q0 + wave * 16 + lanelo;
        const bf16* qp = Q + qoff + (long)row * 128;
        #pragma unroll
        for (int dk = 0; dk < 4; ++dk) {
            int kb = dk * 32 + hi * 8;
            reinterpret_cast<U4*>(&qf[dk])[0] = *reinterpret_cast<const U4*>(qp + kb);
            reinterpret_cast<U4*>(&qf[dk])[1] = *reinterpret_cast<const U4*>(qp + kb + 16);
        }
    }

    v8f oacc[8] = {};
    float m_old[8], l[8];
    #pragma unroll
    for (int r = 0; r < 8; ++r) { m_old[r] = -3.0e38f; l[r] = 0.0f; }

    const int nkt = q0 / 32 + 4;          // causal: keys 0 .. q0+127
#if USE_TDM
    if (wave == 0)
        tdm_load_2d(lds_off(&Ks[0][0]), &K[kvoff], 128, 32, 128ull);
#else
    loadK(0);
    storeK(0);
#endif
    loadV(0);
    storeV(0);
    int cur = 0;

    #pragma unroll 1
    for (int kt = 0; kt < nkt; ++kt) {
#if USE_TDM
        __builtin_amdgcn_s_wait_tensorcnt((short)0);
#endif
        __syncthreads();
        const int kst = kt * 32;
        const bool has_next = (kt + 1 < nkt);
        if (has_next) {
#if USE_TDM
            if (wave == 0)
                tdm_load_2d(lds_off(&Ks[cur ^ 1][0]), &K[kvoff + (long)(kst + 32) * 128],
                            128, 32, 128ull);
#else
            loadK(kst + 32);
#endif
            loadV(kst + 32);              // overlap next tile with compute
        }

        // ---- S = Q K^T (16 rows x 32 keys per wave)
        v8f sv[2] = {};
        #pragma unroll
        for (int dk = 0; dk < 4; ++dk) {
            #pragma unroll
            for (int n2 = 0; n2 < 2; ++n2) {
                v16bf kf;
                const bf16* p = &Ks[cur][(n2 * 16 + lanelo) * 128 + dk * 32 + hi * 8];
                reinterpret_cast<U4*>(&kf)[0] = *reinterpret_cast<const U4*>(p);
                reinterpret_cast<U4*>(&kf)[1] = *reinterpret_cast<const U4*>(p + 16);
                sv[n2] = __builtin_amdgcn_wmma_f32_16x16x32_bf16(
                    false, qf[dk], false, kf, (short)0, sv[n2], false, false);
            }
        }

        // ---- online softmax (row stats live inside one 16-lane half)
        float alpha[8];
        const int pbase = wave * (16 * 32);
        #pragma unroll
        for (int r = 0; r < 8; ++r) {
            const int row_g = q0 + wave * 16 + hi * 8 + r;
            float s0 = sv[0][r] * scale;
            float s1 = sv[1][r] * scale;
            if (kst + lanelo      > row_g) s0 = -3.0e38f;
            if (kst + 16 + lanelo > row_g) s1 = -3.0e38f;
            float mx = fmaxf(s0, s1);
            #pragma unroll
            for (int mk = 8; mk >= 1; mk >>= 1) mx = fmaxf(mx, __shfl_xor(mx, mk, 32));
            const float mn = fmaxf(m_old[r], mx);
            const float al = __expf(m_old[r] - mn);
            const float p0 = __expf(s0 - mn);
            const float p1 = __expf(s1 - mn);
            float sm = p0 + p1;
            #pragma unroll
            for (int mk = 8; mk >= 1; mk >>= 1) sm += __shfl_xor(sm, mk, 32);
            l[r] = l[r] * al + sm;
            m_old[r] = mn;
            alpha[r] = al;
            Ps[pbase + (hi * 8 + r) * 32 + lanelo]      = (bf16)p0;
            Ps[pbase + (hi * 8 + r) * 32 + 16 + lanelo] = (bf16)p1;
        }
        #pragma unroll
        for (int n = 0; n < 8; ++n)
            #pragma unroll
            for (int r = 0; r < 8; ++r)
                oacc[n][r] *= alpha[r];

        // ---- P fragment (wave-private LDS round trip C-layout -> A-layout)
        v16bf pf;
        {
            const bf16* pp = &Ps[pbase + lanelo * 32 + hi * 8];
            reinterpret_cast<U4*>(&pf)[0] = *reinterpret_cast<const U4*>(pp);
            reinterpret_cast<U4*>(&pf)[1] = *reinterpret_cast<const U4*>(pp + 16);
        }
        // ---- O += P * V
        #pragma unroll
        for (int n = 0; n < 8; ++n) {
            v16bf vf;
            const bf16* vp = &Vt[cur][(n * 16 + lanelo) * 32 + hi * 8];
            reinterpret_cast<U4*>(&vf)[0] = *reinterpret_cast<const U4*>(vp);
            reinterpret_cast<U4*>(&vf)[1] = *reinterpret_cast<const U4*>(vp + 16);
            oacc[n] = __builtin_amdgcn_wmma_f32_16x16x32_bf16(
                false, pf, false, vf, (short)0, oacc[n], false, false);
        }

        if (has_next) {
#if !USE_TDM
            storeK(cur ^ 1);
#endif
            storeV(cur ^ 1);
            cur ^= 1;
        }
    }

    // ---- normalize and store to (B,T,H*128) bf16
    #pragma unroll
    for (int r = 0; r < 8; ++r) {
        const int row_g = q0 + wave * 16 + hi * 8 + r;
        const float rinv = 1.0f / l[r];
        const long yb = (((long)b * 2048 + row_g) * 16 + h) * 128;
        #pragma unroll
        for (int n = 0; n < 8; ++n)
            Y[yb + n * 16 + lanelo] = (bf16)(oacc[n][r] * rinv);
    }
}

// ---------------------------------------------------------------------- host
extern "C" void kernel_launch(void* const* d_in, const int* in_sizes, int n_in,
                              void* d_out, int out_size, void* d_ws, size_t ws_size,
                              hipStream_t stream) {
    const float* x    = (const float*)d_in[0];
    const float* cosp = (const float*)d_in[1];
    const float* sinp = (const float*)d_in[2];
    const float* Wq   = (const float*)d_in[3];
    const float* Wk   = (const float*)d_in[4];
    const float* Wv   = (const float*)d_in[5];
    const float* Wo   = (const float*)d_in[6];

    const int B = 4, T = 2048, C = 2048, H = 16, KH = 4, D = 128;
    const long BT = (long)B * T;          // 8192

    char* ws = (char*)d_ws;
    size_t off = 0;
    auto alloc = [&](size_t bytes) { char* p = ws + off; off += (bytes + 255) & ~(size_t)255; return p; };
    bf16*  xb   = (bf16*)alloc((size_t)BT * C * 2);
    bf16*  Wqb  = (bf16*)alloc((size_t)C * C * 2);
    bf16*  Wkb  = (bf16*)alloc((size_t)C * KH * D * 2);
    bf16*  Wvb  = (bf16*)alloc((size_t)C * KH * D * 2);
    bf16*  Wob  = (bf16*)alloc((size_t)C * C * 2);
    float* qraw = (float*)alloc((size_t)BT * C * 4);
    float* kraw = (float*)alloc((size_t)BT * KH * D * 4);
    float* vraw = (float*)alloc((size_t)BT * KH * D * 4);
    bf16*  Qb   = (bf16*)alloc((size_t)BT * C * 2);
    bf16*  Kb   = (bf16*)alloc((size_t)BT * KH * D * 2);
    bf16*  Vb   = (bf16*)alloc((size_t)BT * KH * D * 2);
    bf16*  Yb   = (bf16*)alloc((size_t)BT * C * 2);

    // 1) casts to bf16
    cvt_f32_bf16<<<2048, 256, 0, stream>>>(x,  xb,  BT * C);
    cvt_f32_bf16<<<1024, 256, 0, stream>>>(Wq, Wqb, (long)C * C);
    cvt_f32_bf16<<<512,  256, 0, stream>>>(Wk, Wkb, (long)C * KH * D);
    cvt_f32_bf16<<<512,  256, 0, stream>>>(Wv, Wvb, (long)C * KH * D);
    cvt_f32_bf16<<<1024, 256, 0, stream>>>(Wo, Wob, (long)C * C);

    // 2) QKV projections
    gemm_bf16_wmma<2048, 2048><<<dim3(C / 128, BT / 128), 256, 0, stream>>>(xb, Wqb, qraw, (int)BT);
    gemm_bf16_wmma<512,  2048><<<dim3((KH * D) / 128, BT / 128), 256, 0, stream>>>(xb, Wkb, kraw, (int)BT);
    gemm_bf16_wmma<512,  2048><<<dim3((KH * D) / 128, BT / 128), 256, 0, stream>>>(xb, Wvb, vraw, (int)BT);

    // 3) RoPE + RMSNorm + head-major relayout
    rope_rms<<<(int)(BT * H),  32, 0, stream>>>(qraw, cosp, sinp, Qb, H,  T);
    rope_rms<<<(int)(BT * KH), 32, 0, stream>>>(kraw, cosp, sinp, Kb, KH, T);
    v_relayout<<<(int)(BT * KH), 128, 0, stream>>>(vraw, Vb, KH, T);

    // 4) flash attention
    attn_flash<<<dim3(B * H, T / 128), 256, 0, stream>>>(Qb, Kb, Vb, Yb);

    // 5) output projection -> f32 d_out
    gemm_bf16_wmma<2048, 2048><<<dim3(C / 128, BT / 128), 256, 0, stream>>>(Yb, Wob, (float*)d_out, (int)BT);
}
